// MULTI_46024869543985
// MI455X (gfx1250) — matrix operand order repaired and checked by reading, not compiled, not run on hardware
//
#include <hip/hip_runtime.h>
#include <hip/hip_bf16.h>
#include <math.h>

#define HDIM 64
#define BATCH 256
#define TLEN 256

typedef __attribute__((ext_vector_type(16))) _Float16 v16h;
typedef __attribute__((ext_vector_type(8)))  float    v8f;
typedef __attribute__((ext_vector_type(2)))  _Float16 h2v;
typedef __attribute__((ext_vector_type(4)))  _Float16 h4v;

// ---- workspace byte offsets ----
#define WS_XF16   0                          // [T][B][64] f16        = 8,388,608 B
#define WS_ENCF16 (WS_XF16 + 8388608)        // [B][T][128] f16       = 16,777,216 B
#define WS_WCATE  (WS_ENCF16 + 16777216)     // enc [2][128][256] f16 = 131,072 B
#define WS_WCATD  (WS_WCATE + 131072)        // dec [2][128][256] f16
#define WS_COMBWT (WS_WCATD + 131072)        // [256][64] f16         = 32,768 B
#define WS_BIASE  (WS_COMBWT + 32768)        // [2][256] f32
#define WS_BIASD  (WS_BIASE + 2048)
#define WS_H0     (WS_BIASD + 2048)          // [2][256][64] f32
#define WS_C0     (WS_H0 + 131072)
#define WS_PUB    (WS_C0 + 131072)           // [2 parity][2 dir][256 row][2] f32 = 8192 B
#define WS_CNT    (WS_PUB + 8192)            // grid-barrier counter

__device__ __forceinline__ v8f wmma_f16(v16h a, v16h b, v8f c) {
  return __builtin_amdgcn_wmma_f32_16x16x32_f16(false, a, false, b, (short)0, c,
                                                false, false);
}

// A fragment: 16 x K row-major f16 (row stride in halfs), K-chunk [k0, k0+32)
// Layout: lanes 0-15 -> M=0..15, K=0..15 of chunk ; lanes 16-31 -> K=16..31 (matches B)
// Per lane this is one contiguous 32-byte run -> compiler emits ds_load_b128 pairs.
__device__ __forceinline__ v16h load_a_frag(const _Float16* base, int stride, int k0, int lane) {
  const int m = lane & 15, hh = lane >> 4;
  v16h a;
#pragma unroll
  for (int v = 0; v < 8; ++v) {
    int kk = k0 + hh * 16 + (v << 1);
    a[2 * v]     = base[m * stride + kk];
    a[2 * v + 1] = base[m * stride + kk + 1];
  }
  return a;
}

// B fragment: K x N row-major f16 (row stride = N halfs), rows [k0,k0+32), cols [n0,n0+16)
// Layout (same contiguous k order as A): lanes 0-15 -> K=0..15 of chunk ; lanes 16-31 -> K=16..31
__device__ __forceinline__ v16h load_b_frag(const _Float16* base, int stride, int k0, int n0, int lane) {
  const int n = n0 + (lane & 15), hh = lane >> 4;
  v16h b;
#pragma unroll
  for (int v = 0; v < 8; ++v) {
    int kk = k0 + hh * 16 + (v << 1);
    b[2 * v]     = base[kk * stride + n];
    b[2 * v + 1] = base[(kk + 1) * stride + n];
  }
  return b;
}

// --- fast transcendentals: v_exp_f32 + v_rcp_f32, no IEEE-div sequences --------
__device__ __forceinline__ float fast_rcp(float x) { return __builtin_amdgcn_rcpf(x); }
__device__ __forceinline__ float sigf(float x) {
  // 1/(1+2^(-x*log2e))
  return fast_rcp(1.0f + __builtin_amdgcn_exp2f(x * -1.442695040888963f));
}
__device__ __forceinline__ float tanhf_fast(float x) {
  // tanh(x) = 1 - 2/(1+2^(2x*log2e))
  const float e = __builtin_amdgcn_exp2f(x * 2.885390081777927f);
  return 1.0f - 2.0f * fast_rcp(1.0f + e);
}

__device__ __forceinline__ void grid_barrier(unsigned* cnt, int nblk, int gen) {
  __syncthreads();
  if (threadIdx.x == 0) {
    __threadfence();
    atomicAdd(cnt, 1u);
    const unsigned target = (unsigned)gen * (unsigned)nblk;
    while (atomicAdd(cnt, 0u) < target) { __builtin_amdgcn_s_sleep(1); }
  }
  __syncthreads();
}

// ---------------- prep: f16 transposed weight concat + fused biases -------------
__global__ void prep_kernel(const float* encWih, const float* encWhh,
                            const float* encBih, const float* encBhh,
                            const float* decWih, const float* decWhh,
                            const float* decBih, const float* decBhh,
                            const float* combW, char* ws) {
  const int tid = blockIdx.x * blockDim.x + threadIdx.x;  // 65536 total
  _Float16* wcE = (_Float16*)(ws + WS_WCATE);
  _Float16* wcD = (_Float16*)(ws + WS_WCATD);
  if (tid < 2 * 128 * 256) {
    const int dir = tid >> 15, rem = tid & 32767;
    const int k = rem >> 8, n = rem & 255;
    // rows 0..63 = Wih^T, rows 64..127 = Whh^T : element [k][n] = W[n][k]
    float e = (k < 64) ? encWih[dir * 16384 + n * 64 + k]
                       : encWhh[dir * 16384 + n * 64 + (k - 64)];
    float d = (k < 64) ? decWih[dir * 16384 + n * 64 + k]
                       : decWhh[dir * 16384 + n * 64 + (k - 64)];
    wcE[tid] = (_Float16)e;
    wcD[tid] = (_Float16)d;
  }
  if (tid < 256 * 64) {
    const int k = tid >> 6, n = tid & 63;
    ((_Float16*)(ws + WS_COMBWT))[tid] = (_Float16)combW[n * 256 + k];  // comb_W^T
  }
  if (tid < 512) {
    ((float*)(ws + WS_BIASE))[tid] = encBih[tid] + encBhh[tid];
    ((float*)(ws + WS_BIASD))[tid] = decBih[tid] + decBhh[tid];
  }
  if (tid == 0) *((unsigned*)(ws + WS_CNT)) = 0u;
}

// ---------------- embedding gather -> f16 [T][B][H] -----------------------------
__global__ void embed_kernel(const int* xin, const float* emb, char* ws) {
  const int e = blockIdx.x * blockDim.x + threadIdx.x;  // T*B*H = 4,194,304
  if (e >= TLEN * BATCH * HDIM) return;
  const int t = e >> 14, rem = e & 16383;
  const int b = rem >> 6, h = rem & 63;
  const int idx = xin[b * TLEN + t];
  ((_Float16*)(ws + WS_XF16))[e] = (_Float16)emb[idx * HDIM + h];
}

// ---------------- encoder: 32 WGs = 16 batch-chunks x 2 dirs, persistent --------
__global__ __launch_bounds__(512) void encoder_kernel(char* ws) {
  const int tid = threadIdx.x;
  const int wv = tid >> 5, lane = tid & 31;
  const int dir = blockIdx.x >> 4, chunk = blockIdx.x & 15;
  const int rbase = chunk * 16;

  __shared__ _Float16 sX[16][64];
  __shared__ _Float16 sH[16][64];
  __shared__ float sC[16][64];
  __shared__ float sHf[16][64];
  __shared__ float sG[16][256];

  const _Float16* Xf   = (const _Float16*)(ws + WS_XF16);
  _Float16*       encF = (_Float16*)(ws + WS_ENCF16);
  const _Float16* wcat = (const _Float16*)(ws + WS_WCATE) + dir * 32768;
  const float*    bias = (const float*)(ws + WS_BIASE) + dir * 256;
  float* h0g = (float*)(ws + WS_H0);
  float* c0g = (float*)(ws + WS_C0);

  for (int e = tid; e < 1024; e += 512) {
    const int r = e >> 6, j = e & 63;
    sH[r][j] = (_Float16)0.f;
    sC[r][j] = 0.f;
  }

  // weights for this wave's 16 gate columns, pinned in VGPRs for all 256 steps
  const int n0 = wv * 16;
  const v16h wB0 = load_b_frag(wcat, 256, 0,  n0, lane);   // x part K 0..31
  const v16h wB1 = load_b_frag(wcat, 256, 32, n0, lane);   // x part K 32..63
  const v16h wB2 = load_b_frag(wcat, 256, 64, n0, lane);   // h part K 0..31
  const v16h wB3 = load_b_frag(wcat, 256, 96, n0, lane);   // h part K 32..63
  const float bn = bias[n0 + (lane & 15)];
  __syncthreads();

  for (int tt = 0; tt < TLEN; ++tt) {
    const int t = (dir == 0) ? tt : (TLEN - 1 - tt);
    for (int e = tid; e < 1024; e += 512) {
      const int r = e >> 6, j = e & 63;
      sX[r][j] = Xf[(size_t)t * (BATCH * HDIM) + (rbase + r) * HDIM + j];
    }
    __syncthreads();

    v8f acc = {};
    acc = wmma_f16(load_a_frag(&sX[0][0], 64, 0,  lane), wB0, acc);
    acc = wmma_f16(load_a_frag(&sX[0][0], 64, 32, lane), wB1, acc);
    acc = wmma_f16(load_a_frag(&sH[0][0], 64, 0,  lane), wB2, acc);
    acc = wmma_f16(load_a_frag(&sH[0][0], 64, 32, lane), wB3, acc);
    {
      const int hh = lane >> 4, nn = n0 + (lane & 15);
#pragma unroll
      for (int v = 0; v < 8; ++v) sG[v + 8 * hh][nn] = acc[v] + bn;
    }
    __syncthreads();

    for (int e = tid; e < 1024; e += 512) {
      const int r = e >> 6, j = e & 63;
      const float gi = sG[r][j], gf = sG[r][64 + j], gg = sG[r][128 + j], go = sG[r][192 + j];
      const float c2 = sigf(gf) * sC[r][j] + sigf(gi) * tanhf_fast(gg);
      const float h  = sigf(go) * tanhf_fast(c2);
      sC[r][j]  = c2;
      sHf[r][j] = h;
      sH[r][j]  = (_Float16)h;
      encF[((size_t)(rbase + r) * TLEN + t) * 128 + dir * 64 + j] = (_Float16)h;
    }
    __syncthreads();
  }

  for (int e = tid; e < 1024; e += 512) {
    const int r = e >> 6, j = e & 63;
    h0g[dir * 16384 + (rbase + r) * 64 + j] = sHf[r][j];
    c0g[dir * 16384 + (rbase + r) * 64 + j] = sC[r][j];
  }
}

// ---------------- enc_logits: enc_out[:, -1, :] @ cls1_W^T + b ------------------
__global__ void enc_logits_kernel(const float* cls1W, const float* cls1b,
                                  const char* ws, float* out) {
  const int b = blockIdx.x * blockDim.x + threadIdx.x;
  if (b >= BATCH) return;
  const _Float16* e = (const _Float16*)(ws + WS_ENCF16) + ((size_t)b * TLEN + (TLEN - 1)) * 128;
  for (int k = 0; k < 2; ++k) {
    float acc = cls1b[k];
    for (int j = 0; j < 128; ++j) acc += (float)e[j] * cls1W[k * 128 + j];
    out[b * 2 + k] = acc;
  }
}

// ---------------- decoder: 16 persistent WGs, grid-barrier per step -------------
__global__ __launch_bounds__(512) void decoder_kernel(
    const float* attnW, const float* attnB, const float* combB,
    const float* cls2W, const float* cls2B, char* ws, float* out) {
  const int tid = threadIdx.x;
  const int wv = tid >> 5, lane = tid & 31;
  const int rbase = blockIdx.x * 16;
  const int nblk = gridDim.x;

  __shared__ float    sH[2][16][64];
  __shared__ float    sC[2][16][64];
  __shared__ _Float16 sHh[2][16][64];
  __shared__ _Float16 sIn[16][256];   // [x_di | applied] f16 (A for comb GEMM)
  __shared__ _Float16 sZ[16][64];
  __shared__ float    sSG[16][256];   // attention weights, later gate buffer
  __shared__ float    sAw[128];

  const _Float16* encF   = (const _Float16*)(ws + WS_ENCF16);
  const _Float16* wcat   = (const _Float16*)(ws + WS_WCATD);
  const _Float16* combWT = (const _Float16*)(ws + WS_COMBWT);
  const float* biasD = (const float*)(ws + WS_BIASD);
  const float* h0g   = (const float*)(ws + WS_H0);
  const float* c0g   = (const float*)(ws + WS_C0);
  float*    pub = (float*)(ws + WS_PUB);
  unsigned* cnt = (unsigned*)(ws + WS_CNT);

  // pin cell weights (both directions) in VGPRs
  const int n0 = wv * 16;
  v16h dW[2][4];
#pragma unroll
  for (int d = 0; d < 2; ++d)
#pragma unroll
    for (int kc = 0; kc < 4; ++kc)
      dW[d][kc] = load_b_frag(wcat + d * 32768, 256, kc * 32, n0, lane);
  const float bn0 = biasD[n0 + (lane & 15)];
  const float bn1 = biasD[256 + n0 + (lane & 15)];
  const float aB = attnB[0];

  if (tid < 128) sAw[tid] = attnW[tid];
  for (int e = tid; e < 2048; e += 512) {
    const int d = e >> 10, r = (e >> 6) & 15, j = e & 63;
    const float h = h0g[d * 16384 + (rbase + r) * 64 + j];
    sH[d][r][j]  = h;
    sHh[d][r][j] = (_Float16)h;
    sC[d][r][j]  = c0g[d * 16384 + (rbase + r) * 64 + j];
  }
  __syncthreads();

  // publish initial h partial dots (parity 0): pa = h.w2[0:64], pb = h.w2[64:128]
  if (tid < 32) {
    const int d = tid >> 4, r = tid & 15;
    float pa = 0.f, pb = 0.f;
    for (int j = 0; j < 64; ++j) {
      const float h = sH[d][r][j];
      pa += h * attnW[128 + j];
      pb += h * attnW[192 + j];
    }
    const int row = rbase + r;
    pub[d * 512 + row * 2 + 0] = pa;
    pub[d * 512 + row * 2 + 1] = pb;
  }
  grid_barrier(cnt, nblk, 1);

  for (int t = 0; t < TLEN; ++t) {
    const int par = t & 1;
    const int r = wv;               // wave <-> batch row
    const int row = rbase + r;

    // dh = view(-1,128) of [2,B,H]: row b<128 -> (h0[2b], h0[2b+1]); b>=128 -> h1
    float s2;
    if (row < 128) {
      s2 = pub[par * 1024 + (2 * row) * 2 + 0] +
           pub[par * 1024 + (2 * row + 1) * 2 + 1];
    } else {
      const int bp = row - 128;
      s2 = pub[par * 1024 + 512 + (2 * bp) * 2 + 0] +
           pub[par * 1024 + 512 + (2 * bp + 1) * 2 + 1];
    }

    // scores + softmax, one wave per batch row (8 timesteps per lane)
    const h2v* erow = (const h2v*)(encF + (size_t)row * TLEN * 128);
    float sc[8];
    float mx = -1e30f;
#pragma unroll
    for (int it = 0; it < 8; ++it) {
      const int ts = lane + 32 * it;
      const h2v* p = erow + ts * 64;
      float acc = aB + s2;
#pragma unroll 16
      for (int j = 0; j < 64; ++j) {
        const h2v v2 = p[j];
        acc += (float)v2.x * sAw[2 * j] + (float)v2.y * sAw[2 * j + 1];
      }
      sc[it] = acc;
      mx = fmaxf(mx, acc);
    }
#pragma unroll
    for (int m = 16; m >= 1; m >>= 1) mx = fmaxf(mx, __shfl_xor(mx, m, 32));
    float sum = 0.f;
#pragma unroll
    for (int it = 0; it < 8; ++it) { sc[it] = __expf(sc[it] - mx); sum += sc[it]; }
#pragma unroll
    for (int m = 16; m >= 1; m >>= 1) sum += __shfl_xor(sum, m, 32);
    const float inv = fast_rcp(sum);
#pragma unroll
    for (int it = 0; it < 8; ++it) sSG[r][lane + 32 * it] = sc[it] * inv;

    // applied context + build A-matrix [x_di | applied] (lane owns 4 channels)
    {
      float a0 = 0.f, a1 = 0.f, a2 = 0.f, a3 = 0.f;
      const h4v* pr = (const h4v*)(encF + (size_t)row * TLEN * 128);
      for (int ts = 0; ts < TLEN; ++ts) {
        const float w = sSG[r][ts];
        const h4v v4 = pr[ts * 32 + lane];
        a0 += w * (float)v4.x; a1 += w * (float)v4.y;
        a2 += w * (float)v4.z; a3 += w * (float)v4.w;
      }
      ((h4v*)&sIn[r][0])[lane] = pr[t * 32 + lane];   // x_di = enc_out[:, t, :]
      h4v av;
      av.x = (_Float16)a0; av.y = (_Float16)a1; av.z = (_Float16)a2; av.w = (_Float16)a3;
      ((h4v*)&sIn[r][128])[lane] = av;
    }
    __syncthreads();

    // comb: z = relu([x_di|applied] @ comb_W^T + b)   (4 waves, K=256)
    if (wv < 4) {
      const int zn0 = wv * 16;
      v8f acc = {};
#pragma unroll
      for (int kc = 0; kc < 8; ++kc) {
        const v16h a = load_a_frag(&sIn[0][0], 256, kc * 32, lane);
        const v16h b = load_b_frag(combWT, 64, kc * 32, zn0, lane);
        acc = wmma_f16(a, b, acc);
      }
      const int hh = lane >> 4, nn = zn0 + (lane & 15);
      const float cb = combB[nn];
#pragma unroll
      for (int v = 0; v < 8; ++v) {
        float z = acc[v] + cb;
        sZ[v + 8 * hh][nn] = (_Float16)(z > 0.f ? z : 0.f);
      }
    }
    __syncthreads();

    // two LSTM cells (fwd/bwd) on input z, one gate tile per wave per dir
#pragma unroll
    for (int d = 0; d < 2; ++d) {
      v8f acc = {};
      acc = wmma_f16(load_a_frag(&sZ[0][0],     64, 0,  lane), dW[d][0], acc);
      acc = wmma_f16(load_a_frag(&sZ[0][0],     64, 32, lane), dW[d][1], acc);
      acc = wmma_f16(load_a_frag(&sHh[d][0][0], 64, 0,  lane), dW[d][2], acc);
      acc = wmma_f16(load_a_frag(&sHh[d][0][0], 64, 32, lane), dW[d][3], acc);
      {
        const int hh = lane >> 4, nn = n0 + (lane & 15);
        const float bb = (d == 0) ? bn0 : bn1;
#pragma unroll
        for (int v = 0; v < 8; ++v) sSG[v + 8 * hh][nn] = acc[v] + bb;
      }
      __syncthreads();
      for (int e = tid; e < 1024; e += 512) {
        const int rr = e >> 6, j = e & 63;
        const float gi = sSG[rr][j], gf = sSG[rr][64 + j];
        const float gg = sSG[rr][128 + j], go = sSG[rr][192 + j];
        const float c2 = sigf(gf) * sC[d][rr][j] + sigf(gi) * tanhf_fast(gg);
        const float h  = sigf(go) * tanhf_fast(c2);
        sC[d][rr][j]  = c2;
        sH[d][rr][j]  = h;
        sHh[d][rr][j] = (_Float16)h;
      }
      __syncthreads();
    }

    // dec_logits for this step + publish next-step h dot-products (other parity)
    if (tid < 32) {
      const int rr = tid >> 1, k = tid & 1;
      float acc = cls2B[k];
      for (int j = 0; j < 64; ++j) {
        acc += sH[0][rr][j] * cls2W[k * 128 + j];
        acc += sH[1][rr][j] * cls2W[k * 128 + 64 + j];
      }
      out[512 + ((size_t)(rbase + rr) * TLEN + t) * 2 + k] = acc;
    } else if (tid < 64) {
      const int q = tid - 32;
      const int d = q >> 4, rr = q & 15;
      float pa = 0.f, pb = 0.f;
      for (int j = 0; j < 64; ++j) {
        const float h = sH[d][rr][j];
        pa += h * attnW[128 + j];
        pb += h * attnW[192 + j];
      }
      const int row2 = rbase + rr;
      const int np = (t + 1) & 1;
      pub[np * 1024 + d * 512 + row2 * 2 + 0] = pa;
      pub[np * 1024 + d * 512 + row2 * 2 + 1] = pb;
    }
    grid_barrier(cnt, nblk, t + 2);
  }
}

extern "C" void kernel_launch(void* const* d_in, const int* in_sizes, int n_in,
                              void* d_out, int out_size, void* d_ws, size_t ws_size,
                              hipStream_t stream) {
  (void)in_sizes; (void)n_in; (void)out_size; (void)ws_size;
  const int*   xin    = (const int*)d_in[0];
  const float* emb    = (const float*)d_in[1];
  const float* encWih = (const float*)d_in[2];
  const float* encWhh = (const float*)d_in[3];
  const float* encBih = (const float*)d_in[4];
  const float* encBhh = (const float*)d_in[5];
  const float* decWih = (const float*)d_in[6];
  const float* decWhh = (const float*)d_in[7];
  const float* decBih = (const float*)d_in[8];
  const float* decBhh = (const float*)d_in[9];
  const float* attnW  = (const float*)d_in[10];
  const float* attnB  = (const float*)d_in[11];
  const float* combW  = (const float*)d_in[12];
  const float* combB  = (const float*)d_in[13];
  const float* cls1W  = (const float*)d_in[14];
  const float* cls1b  = (const float*)d_in[15];
  const float* cls2W  = (const float*)d_in[16];
  const float* cls2b  = (const float*)d_in[17];
  char*  ws  = (char*)d_ws;
  float* out = (float*)d_out;

  prep_kernel<<<256, 256, 0, stream>>>(encWih, encWhh, encBih, encBhh,
                                       decWih, decWhh, decBih, decBhh, combW, ws);
  embed_kernel<<<(TLEN * BATCH * HDIM + 255) / 256, 256, 0, stream>>>(xin, emb, ws);
  encoder_kernel<<<32, 512, 0, stream>>>(ws);
  enc_logits_kernel<<<1, 256, 0, stream>>>(cls1W, cls1b, ws, out);
  decoder_kernel<<<16, 512, 0, stream>>>(attnW, attnB, combB, cls2W, cls2b, ws, out);
}
